// EmotionNet_8246337208792
// MI455X (gfx1250) — compile-verified, matrix operand order checked by reading, not run on hardware
//
#include <hip/hip_runtime.h>

// ---------------------------------------------------------------------------
// Two-layer LSTM (B=2048, T=50, D=256, H=512) + linear[2, T*H] + softmax
// bf16 WMMA (v_wmma_f32_16x16x32_bf16) GEMMs with f32 accumulation.
// Global->LDS staging via Tensor Data Mover (tensor_load_to_lds) where the
// source is bf16, double-buffered and pipelined against WMMA via TENSORcnt.
// ---------------------------------------------------------------------------

#define B_   2048
#define T_   50
#define D_   256
#define H_   512
#define NG_  2048          // 4*H
#define MT   128           // M tile (batch rows) per block
#define JT   32            // hidden columns per block (x4 gates = 128 N cols)
#define KC   32            // K chunk per WMMA step
#define LSTR 40            // LDS row stride in bf16 elems (32 + 8 pad)

typedef __attribute__((ext_vector_type(16))) __bf16 v16bf;
typedef __attribute__((ext_vector_type(8)))  float  v8f;
typedef __attribute__((ext_vector_type(4)))  unsigned int v4u;
typedef __attribute__((ext_vector_type(8)))  int v8i;
typedef __attribute__((ext_vector_type(4)))  int v4i;

#if defined(__has_builtin)
#  if __has_builtin(__builtin_amdgcn_tensor_load_to_lds) && __has_builtin(__builtin_amdgcn_s_wait_tensorcnt)
#    define HAVE_TDM 1
#  else
#    define HAVE_TDM 0
#  endif
#else
#  define HAVE_TDM 0
#endif

#if HAVE_TDM
#  if __has_include(<hip/amd_detail/amd_gfx1250_TDM.h>)
     // therock-10.0 headers -> 6-arg builtin (extra int32x8 group)
#    define TDM_CALL(g0, g1, gz) \
        __builtin_amdgcn_tensor_load_to_lds(g0, g1, gz, gz, (v8i){0,0,0,0,0,0,0,0}, 0)
#  else
     // ROCm 7.2 clang-22 -> 5-arg builtin
#    define TDM_CALL(g0, g1, gz) \
        __builtin_amdgcn_tensor_load_to_lds(g0, g1, gz, gz, 0)
#  endif
#endif

struct alignas(16) U4 { unsigned int a, b, c, d; };
struct alignas(16) F4 { float x, y, z, w; };
struct U4x2 { U4 lo, hi; };
union ABFrag { U4x2 q; v16bf v; };

__device__ __forceinline__ unsigned short f2bf(float f) {
    unsigned int u = __float_as_uint(f);
    unsigned int r = u + 0x7FFFu + ((u >> 16) & 1u);   // round-to-nearest-even
    return (unsigned short)(r >> 16);
}
__device__ __forceinline__ float bf2f(unsigned short s) {
    return __uint_as_float(((unsigned int)s) << 16);
}
__device__ __forceinline__ unsigned int pack2(float lo, float hi) {
    return (unsigned int)f2bf(lo) | ((unsigned int)f2bf(hi) << 16);
}
__device__ __forceinline__ float sigm(float v)  { return 1.f / (1.f + __expf(-v)); }
__device__ __forceinline__ float tanhft(float v){ return 2.f / (1.f + __expf(-2.f * v)) - 1.f; }

// ---- vector staging helpers (256 threads stage a 128x32 bf16 tile) --------
__device__ __forceinline__ void stage_f32(unsigned short* lds,
                                          const float* __restrict__ src,
                                          size_t rowStride, int mbase, int col0,
                                          int srow, int shalf) {
    const F4* p = (const F4*)(src + (size_t)(mbase + srow) * rowStride + col0 + shalf * 16);
    __builtin_prefetch(p + 8, 0, 1);   // next K chunk of this x row
    F4 f0 = p[0], f1 = p[1], f2 = p[2], f3 = p[3];
    U4 u0, u1;
    u0.a = pack2(f0.x, f0.y); u0.b = pack2(f0.z, f0.w);
    u0.c = pack2(f1.x, f1.y); u0.d = pack2(f1.z, f1.w);
    u1.a = pack2(f2.x, f2.y); u1.b = pack2(f2.z, f2.w);
    u1.c = pack2(f3.x, f3.y); u1.d = pack2(f3.z, f3.w);
    *(U4*)&lds[srow * LSTR + shalf * 16]     = u0;
    *(U4*)&lds[srow * LSTR + shalf * 16 + 8] = u1;
}

#if !HAVE_TDM
__device__ __forceinline__ void stage_bf16(unsigned short* lds,
                                           const unsigned short* __restrict__ src,
                                           size_t rowStride, int mbase, int col0,
                                           int srow, int shalf) {
    const U4* p = (const U4*)(src + (size_t)(mbase + srow) * rowStride + col0 + shalf * 16);
    U4 u0 = p[0], u1 = p[1];
    *(U4*)&lds[srow * LSTR + shalf * 16]     = u0;
    *(U4*)&lds[srow * LSTR + shalf * 16 + 8] = u1;
}
__device__ __forceinline__ void stage_W(unsigned short* lds,
                                        const unsigned short* __restrict__ Wc,
                                        int K, int j0, int k0, int srow, int shalf) {
    const int gate = srow >> 5;
    const int jj   = srow & 31;
    const int n    = gate * H_ + j0 + jj;
    const U4* p = (const U4*)(Wc + (size_t)n * K + k0 + shalf * 16);
    U4 u0 = p[0], u1 = p[1];
    __builtin_prefetch(p + 4, 0, 1);
    *(U4*)&lds[srow * LSTR + shalf * 16]     = u0;
    *(U4*)&lds[srow * LSTR + shalf * 16 + 8] = u1;
}
#endif

#if HAVE_TDM
// ---------------------------------------------------------------------------
// TDM: DMA a (tileRows x 32) bf16 tile (row stride = rowStride elems) into LDS
// at byte offset ldsOff, inserting 16B of padding every 64B so the LDS layout
// matches LSTR=40 bf16/row. ISA D# encoding per CDNA5 ISA.
// ---------------------------------------------------------------------------
__device__ __forceinline__ void tdm_tile(const unsigned short* gsrc, unsigned ldsOff,
                                         int rowStride, int tensorRows, int tileRows) {
    const unsigned long long ga = (unsigned long long)(size_t)gsrc;
    v4u g0;
    g0.x = 1u;                                    // count=1, user mode, no gather
    g0.y = ldsOff;                                // lds_addr (bytes)
    g0.z = (unsigned)ga;                          // global_addr[31:0]
    g0.w = (unsigned)((ga >> 32) & 0x1FFFFFFull)  // global_addr[56:32]
         | (2u << 30);                            // type=2 ("image")
    v8i g1;
    g1[0] = (1 << 16)     // data_size = 2 bytes (bf16)
          | (1 << 20)     // pad_enable
          | (3 << 22)     // pad_interval: 16 DWORDs (= one 64B tile row)
          | (3 << 25);    // pad_amount: 4 DWORDs (= 16B -> 80B LDS row stride)
    g1[1] = (rowStride & 0xFFFF) << 16;                               // tensor_dim0 lo
    g1[2] = ((rowStride >> 16) & 0xFFFF) | ((tensorRows & 0xFFFF) << 16);
    g1[3] = ((tensorRows >> 16) & 0xFFFF) | (KC << 16);               // tile_dim0 = 32
    g1[4] = tileRows & 0xFFFF;                                        // tile_dim1 (dim2=0)
    g1[5] = rowStride;                                                // dim0_stride lo32
    g1[6] = 0;
    g1[7] = 0;
    const v4i gz = (v4i){0, 0, 0, 0};
    TDM_CALL(g0, g1, gz);
}
#endif

// ---------------------------------------------------------------------------
// Fused step kernel: gates = [A_in | h_prev] @ Wc^T + bias ; LSTM cell update.
// Grid: (16, 16) = (M tiles, J tiles). Block: 256 threads = 8 waves.
// Wave w computes rows [w*16, w*16+16) x all 128 N cols (8 wmma accumulators).
// ---------------------------------------------------------------------------
__global__ __launch_bounds__(256)
void lstm_step(const float* __restrict__ x,            // nullable: [B][T][D] f32
               const unsigned short* __restrict__ ain, // bf16 input acts when x==null
               int Kin,
               const unsigned short* __restrict__ hprev, // [B][H] bf16
               const unsigned short* __restrict__ Wc,    // [NG][K] bf16
               const float* __restrict__ bias,           // [NG] (b_ih + b_hh)
               float* __restrict__ cstate,               // [B][H] f32 (in place)
               unsigned short* __restrict__ hnext,       // [B][H] bf16
               int t, int K) {
    __shared__ unsigned short lA[2][MT * LSTR];
    __shared__ unsigned short lB[2][MT * LSTR];

    const int tid   = threadIdx.x;
    const int lane  = tid & 31;
    const int wave  = tid >> 5;
    const int mbase = blockIdx.x * MT;
    const int j0    = blockIdx.y * JT;
    const int srow  = tid >> 1;
    const int shalf = tid & 1;
    const int nchunks = K / KC;

    v8f acc[8];
#pragma unroll
    for (int i = 0; i < 8; ++i) acc[i] = (v8f){0.f, 0.f, 0.f, 0.f, 0.f, 0.f, 0.f, 0.f};

#if HAVE_TDM
    // wave 0 issues TDM descriptors; all operands are wave-uniform (SGPRs).
    #define ISSUE_CHUNK(kcv)                                                          \
        do {                                                                          \
            const int k0_ = (kcv) * KC;                                               \
            const int buf_ = (kcv) & 1;                                               \
            const unsigned ldsA_ = (unsigned)(size_t)&lA[buf_][0];                    \
            const unsigned ldsB_ = (unsigned)(size_t)&lB[buf_][0];                    \
            if (k0_ < Kin) {                                                          \
                if (!x) tdm_tile(ain + (size_t)mbase * Kin + k0_, ldsA_, Kin, B_, MT);\
            } else {                                                                  \
                tdm_tile(hprev + (size_t)mbase * H_ + (k0_ - Kin), ldsA_, H_, B_, MT);\
            }                                                                         \
            for (int g_ = 0; g_ < 4; ++g_)                                            \
                tdm_tile(Wc + (size_t)(g_ * H_ + j0) * K + k0_,                       \
                         ldsB_ + (unsigned)(g_ * 32 * LSTR * 2), K, NG_, 32);         \
        } while (0)

    if (wave == 0) ISSUE_CHUNK(0);
#endif

    for (int kc = 0; kc < nchunks; ++kc) {
        const int k0  = kc * KC;
        const int cur = kc & 1;

        // f32 x-portion of layer 1: convert-stage with vector loads
        if (x && k0 < Kin)
            stage_f32(&lA[cur][0], x + (size_t)t * D_, (size_t)T_ * D_, mbase, k0, srow, shalf);

#if HAVE_TDM
        if (wave == 0) {
            if (kc + 1 < nchunks) {
                ISSUE_CHUNK(kc + 1);
                // in-order TDM completion: outstanding <= ops(kc+1) means chunk kc landed
                const bool nextHasA = !(x && (kc + 1) * KC < Kin);
                if (nextHasA) __builtin_amdgcn_s_wait_tensorcnt((short)5);
                else          __builtin_amdgcn_s_wait_tensorcnt((short)4);
            } else {
                __builtin_amdgcn_s_wait_tensorcnt((short)0);
            }
        }
#else
        if (k0 < Kin) {
            if (!x) stage_bf16(&lA[cur][0], ain, (size_t)Kin, mbase, k0, srow, shalf);
        } else {
            stage_bf16(&lA[cur][0], hprev, (size_t)H_, mbase, k0 - Kin, srow, shalf);
        }
        stage_W(&lB[cur][0], Wc, K, j0, k0, srow, shalf);
#endif
        __syncthreads();

        // ---- A fragment (16x32 bf16): lane&15 = M row, lane>>4 selects K half
        ABFrag af;
        {
            const int row = wave * 16 + (lane & 15);
            const int kh  = lane >> 4;
            af.q.lo = *(const U4*)&lA[cur][row * LSTR + kh * 8];
            af.q.hi = *(const U4*)&lA[cur][row * LSTR + 16 + kh * 8];
        }
#pragma unroll
        for (int nt = 0; nt < 8; ++nt) {
            ABFrag bfq;
            const int brow = nt * 16 + (lane & 15);
            const int kh   = lane >> 4;
            bfq.q.lo = *(const U4*)&lB[cur][brow * LSTR + kh * 8];
            bfq.q.hi = *(const U4*)&lB[cur][brow * LSTR + 16 + kh * 8];
            acc[nt] = __builtin_amdgcn_wmma_f32_16x16x32_bf16(
                false, af.v, false, bfq.v, (short)0, acc[nt], false, false);
        }
        __syncthreads();
    }

    // ---- epilogue: i/f/g/o for (row, j) are in accumulators nt = gate*2+jn,
    // same lane & same vector element -> fully local LSTM cell update.
    const int mrow0 = mbase + wave * 16;
    const int lhalf = (lane >> 4) << 3;   // +8 row offset for lanes 16..31
#pragma unroll
    for (int jn = 0; jn < 2; ++jn) {
        const int j   = j0 + jn * 16 + (lane & 15);
        const float bi  = bias[0 * H_ + j];
        const float bfv = bias[1 * H_ + j];
        const float bg  = bias[2 * H_ + j];
        const float bo  = bias[3 * H_ + j];
#pragma unroll
        for (int r = 0; r < 8; ++r) {
            const int row = mrow0 + r + lhalf;
            const float iv = sigm  (acc[0 * 2 + jn][r] + bi);
            const float fv = sigm  (acc[1 * 2 + jn][r] + bfv);
            const float gv = tanhft(acc[2 * 2 + jn][r] + bg);
            const float ov = sigm  (acc[3 * 2 + jn][r] + bo);
            const size_t off = (size_t)row * H_ + j;
            const float cn = fv * cstate[off] + iv * gv;
            cstate[off] = cn;
            hnext[off]  = f2bf(ov * tanhft(cn));
        }
    }
}

// ---------------------------------------------------------------------------
// Per-step logits accumulation: logits[b][c] += h2_t[b] . W_lin[c][t*H : +H]
// ---------------------------------------------------------------------------
__global__ __launch_bounds__(256)
void logits_acc(const unsigned short* __restrict__ h2,
                const float* __restrict__ Wlin,
                float* __restrict__ logits, int t) {
    const int lane = threadIdx.x & 31;
    const int wave = threadIdx.x >> 5;
    const int row  = blockIdx.x * 8 + wave;
    const float* w0 = Wlin + (size_t)t * H_ + lane * 16;
    const float* w1 = Wlin + (size_t)T_ * H_ + (size_t)t * H_ + lane * 16;
    const unsigned short* hp = h2 + (size_t)row * H_ + lane * 16;
    float s0 = 0.f, s1 = 0.f;
#pragma unroll
    for (int u = 0; u < 16; ++u) {
        const float hv = bf2f(hp[u]);
        s0 += hv * w0[u];
        s1 += hv * w1[u];
    }
    for (int off = 16; off > 0; off >>= 1) {
        s0 += __shfl_xor(s0, off, 32);
        s1 += __shfl_xor(s1, off, 32);
    }
    if (lane == 0) {
        logits[row * 2 + 0] += s0;   // single owner per row; stream-ordered
        logits[row * 2 + 1] += s1;
    }
}

// ---------------------------------------------------------------------------
// Prep: fuse [W_ih | W_hh] into bf16 Wc matrices.
// ---------------------------------------------------------------------------
__global__ void prep_weights(const float* __restrict__ Wih1, const float* __restrict__ Whh1,
                             const float* __restrict__ Wih2, const float* __restrict__ Whh2,
                             unsigned short* __restrict__ Wc1, unsigned short* __restrict__ Wc2) {
    const size_t stride = (size_t)gridDim.x * blockDim.x;
    const size_t idx = (size_t)blockIdx.x * blockDim.x + threadIdx.x;
    const size_t n1 = (size_t)NG_ * 768;
    const size_t n2 = (size_t)NG_ * 1024;
    for (size_t i = idx; i < n1; i += stride) {
        const size_t n = i / 768, k = i % 768;
        const float v = (k < 256) ? Wih1[n * 256 + k] : Whh1[n * 512 + (k - 256)];
        Wc1[i] = f2bf(v);
    }
    for (size_t i = idx; i < n2; i += stride) {
        const size_t n = i / 1024, k = i % 1024;
        const float v = (k < 512) ? Wih2[n * 512 + k] : Whh2[n * 512 + (k - 512)];
        Wc2[i] = f2bf(v);
    }
}

__global__ void init_state(unsigned short* __restrict__ h1, unsigned short* __restrict__ h2,
                           float* __restrict__ c1, float* __restrict__ c2,
                           float* __restrict__ logits,
                           float* __restrict__ bias1, float* __restrict__ bias2,
                           const float* __restrict__ bih1, const float* __restrict__ bhh1,
                           const float* __restrict__ bih2, const float* __restrict__ bhh2) {
    const size_t stride = (size_t)gridDim.x * blockDim.x;
    const size_t idx = (size_t)blockIdx.x * blockDim.x + threadIdx.x;
    const size_t NS = (size_t)B_ * H_;
    for (size_t i = idx; i < NS; i += stride) {
        h1[i] = 0; h1[i + NS] = 0;
        h2[i] = 0; h2[i + NS] = 0;
        c1[i] = 0.f; c2[i] = 0.f;
    }
    for (size_t i = idx; i < (size_t)NG_; i += stride) {
        bias1[i] = bih1[i] + bhh1[i];
        bias2[i] = bih2[i] + bhh2[i];
    }
    for (size_t i = idx; i < (size_t)B_ * 2; i += stride) logits[i] = 0.f;
}

__global__ void softmax_out(const float* __restrict__ logits, const float* __restrict__ blin,
                            float* __restrict__ out) {
    const int b = blockIdx.x * blockDim.x + threadIdx.x;
    if (b >= B_) return;
    const float l0 = logits[b * 2 + 0] + blin[0];
    const float l1 = logits[b * 2 + 1] + blin[1];
    const float m  = fmaxf(l0, l1);
    const float e0 = __expf(l0 - m), e1 = __expf(l1 - m);
    const float inv = 1.f / (e0 + e1);
    out[b * 2 + 0] = e0 * inv;
    out[b * 2 + 1] = e1 * inv;
}

// ---------------------------------------------------------------------------
extern "C" void kernel_launch(void* const* d_in, const int* in_sizes, int n_in,
                              void* d_out, int out_size, void* d_ws, size_t ws_size,
                              hipStream_t stream) {
    const float* x    = (const float*)d_in[0];
    const float* Wih1 = (const float*)d_in[1];
    const float* Whh1 = (const float*)d_in[2];
    const float* bih1 = (const float*)d_in[3];
    const float* bhh1 = (const float*)d_in[4];
    const float* Wih2 = (const float*)d_in[5];
    const float* Whh2 = (const float*)d_in[6];
    const float* bih2 = (const float*)d_in[7];
    const float* bhh2 = (const float*)d_in[8];
    const float* Wlin = (const float*)d_in[9];
    const float* blin = (const float*)d_in[10];
    (void)in_sizes; (void)n_in; (void)out_size; (void)ws_size;

    // ---- workspace carve-up (~24 MB) ----
    char* ws = (char*)d_ws;
    unsigned short* Wc1 = (unsigned short*)ws;  ws += (size_t)NG_ * 768 * 2;
    unsigned short* Wc2 = (unsigned short*)ws;  ws += (size_t)NG_ * 1024 * 2;
    float* bias1 = (float*)ws;                  ws += (size_t)NG_ * 4;
    float* bias2 = (float*)ws;                  ws += (size_t)NG_ * 4;
    const size_t SZ = (size_t)B_ * H_;          // elems per h buffer
    unsigned short* h1 = (unsigned short*)ws;   ws += 2 * SZ * 2;  // ping-pong
    unsigned short* h2 = (unsigned short*)ws;   ws += 2 * SZ * 2;  // ping-pong
    float* c1 = (float*)ws;                     ws += SZ * 4;
    float* c2 = (float*)ws;                     ws += SZ * 4;
    float* logits = (float*)ws;                 ws += (size_t)B_ * 2 * 4;

    prep_weights<<<512, 256, 0, stream>>>(Wih1, Whh1, Wih2, Whh2, Wc1, Wc2);
    init_state<<<2048, 256, 0, stream>>>(h1, h2, c1, c2, logits, bias1, bias2,
                                         bih1, bhh1, bih2, bhh2);

    dim3 grid(B_ / MT, H_ / JT);   // (16, 16)
    int p = 0;
    for (int t = 0; t < T_; ++t) {
        // layer 1: A = [x_t (256) | h1_prev (512)], K = 768
        lstm_step<<<grid, 256, 0, stream>>>(x, nullptr, D_, h1 + p * SZ, Wc1, bias1,
                                            c1, h1 + (1 - p) * SZ, t, 768);
        // layer 2: A = [h1_new (512) | h2_prev (512)], K = 1024
        lstm_step<<<grid, 256, 0, stream>>>(nullptr, h1 + (1 - p) * SZ, H_, h2 + p * SZ,
                                            Wc2, bias2, c2, h2 + (1 - p) * SZ, t, 1024);
        logits_acc<<<256, 256, 0, stream>>>(h2 + (1 - p) * SZ, Wlin, logits, t);
        p ^= 1;
    }
    softmax_out<<<8, 256, 0, stream>>>(logits, blin, (float*)d_out);
}
